// RNN_spike_24489903522058
// MI455X (gfx1250) — matrix operand order repaired and checked by reading, not compiled, not run on hardware
//
#include <hip/hip_runtime.h>
#include <cstdint>

#define B_TOT   512
#define C_CH    3
#define T_STEPS 500
#define D_IN    40
#define N_HID   256
#define N_OUT   12
#define KIN     120   // C*D
#define KINP    128   // padded K for layer 1

#define W1S_HALVES (16 * 4 * 32 * 16)                 // swizzled W1: 64 KB of bf16
#define W1S_BYTES  (W1S_HALVES * 2)

typedef __attribute__((ext_vector_type(16))) __bf16 v16bf;
typedef __attribute__((ext_vector_type(8)))  float  v8f;

__device__ __forceinline__ uint32_t pk2(float a, float b) {
  return (uint32_t)__builtin_bit_cast(unsigned short, (__bf16)a) |
         ((uint32_t)__builtin_bit_cast(unsigned short, (__bf16)b) << 16);
}

__device__ __forceinline__ v8f wmma_bf16(v16bf a, v16bf b, v8f c) {
  // emits v_wmma_f32_16x16x32_bf16
  return __builtin_amdgcn_wmma_f32_16x16x32_bf16(false, a, false, b, (short)0, c,
                                                 false, false);
}

// Branchless streamed B-fragment of W1 (fp32 [n][120], L2-resident).
// Only the (kk==3, v>=4) elements can be out of range (compile-time known); those
// load a clamped in-bounds address and select 0 -> no exec-mask churn, loads clause.
__device__ __forceinline__ v16bf load_b_w1_stream(const float* __restrict__ W1,
                                                  int n, int kk, int h) {
  v16bf f;
#pragma unroll
  for (int v = 0; v < 8; ++v) {
    const int khi = 32 * kk + 16 + 2 * v;        // worst case (h=1), compile-time
    if (khi < KIN) {
      const int k = 32 * kk + h * 16 + 2 * v;
      const float2 q = *(const float2*)(W1 + n * KIN + k);
      f[2 * v] = (__bf16)q.x; f[2 * v + 1] = (__bf16)q.y;
    } else {
      const int k0 = 32 * kk + 2 * v;            // h=0 address, always in bounds
      const float2 q = *(const float2*)(W1 + n * KIN + k0);
      f[2 * v]     = (h == 0) ? (__bf16)q.x : (__bf16)0.f;
      f[2 * v + 1] = (h == 0) ? (__bf16)q.y : (__bf16)0.f;
    }
  }
  return f;
}

// Prep: swizzle W1 (fp32 [256][120]) into zero-padded bf16 B-fragment order in global ws.
//   idx = ((ntile*4 + kk)*32 + lane)*16 + j ;  K = kk*32 + (lane>>4)*16 + j ; N = ntile*16 + (lane&15)
__global__ __launch_bounds__(256)
void w1_swz_kernel(const float* __restrict__ W1, unsigned short* __restrict__ W1S) {
  const int i = blockIdx.x * 256 + threadIdx.x;  // u32 pair index, 16384 total
  if (i >= W1S_HALVES / 2) return;
  const int idx = 2 * i;
  const int j = idx & 15, lp = (idx >> 4) & 31, kk = (idx >> 9) & 3, ntl = idx >> 11;
  const int n = ntl * 16 + (lp & 15);
  const int k = kk * 32 + (lp >> 4) * 16 + j;
  uint32_t u = 0u;
  if (k < KIN) { const float2 q = *(const float2*)(W1 + n * KIN + k); u = pk2(q.x, q.y); }
  ((uint32_t*)W1S)[i] = u;
}

template <bool SWZ>
__global__ __launch_bounds__(256)
void alif_rnn_kernel(const float* __restrict__ X,   const float* __restrict__ THR,
                     const float* __restrict__ W1,  const float* __restrict__ B1,
                     const float* __restrict__ W2IN,const float* __restrict__ B2IN,
                     const float* __restrict__ W2RC,const float* __restrict__ B2RC,
                     const float* __restrict__ W3,  const float* __restrict__ B3,
                     const float* __restrict__ TAD1,const float* __restrict__ TM1,
                     const float* __restrict__ TAD2,const float* __restrict__ TM2,
                     const float* __restrict__ TM3, float* __restrict__ OUT,
                     const unsigned short* __restrict__ W1S)
{
  // All WMMA-consumed arrays stored pre-swizzled in fragment order:
  //  B-side: idx = ((ntile*8 + kk)*32 + lane)*16 + j   A-side: idx = (kk*32 + lane)*16 + j
  // -> every fragment load is one contiguous 32-byte LDS read (2x ds_load_b128).
  __shared__ alignas(32) unsigned short lw2i[16 * 8 * 32 * 16]; // 128 KB: W2_in  bf16
  __shared__ alignas(32) unsigned short lw2r[16 * 8 * 32 * 16]; // 128 KB: W2_rec bf16
  __shared__ alignas(32) unsigned short lw3 [8 * 32 * 16];      //   8 KB: W3 (rows 12..15 = 0)
  __shared__ alignas(32) unsigned short ls1 [8 * 32 * 16];      //   8 KB: s1 spikes, A-layout
  __shared__ alignas(32) unsigned short ls2 [8 * 32 * 16];      //   8 KB: s2 spikes, A-layout
  __shared__ alignas(32) unsigned short lxt [4 * 32 * 16];      //   4 KB: input spikes, A-layout

  const int tid  = threadIdx.x;
  const int lane = tid & 31, wave = tid >> 5;
  const int lm   = lane & 15, h = lane >> 4;
  const int b0   = blockIdx.x * 16;
  const float thr = THR[0];

  // ---- one-time staging: W2_in / W2_rec / W3 -> LDS bf16, swizzled to fragment order
  for (int i = tid; i < 32768; i += 256) {          // u32 pairs (j even -> k,k+1 contiguous)
    const int idx = 2 * i;
    const int j = idx & 15, lp = (idx >> 4) & 31, kk = (idx >> 9) & 7, ntl = idx >> 12;
    const int n = ntl * 16 + (lp & 15);
    const int k = kk * 32 + (lp >> 4) * 16 + j;
    const float2 qi = *(const float2*)(W2IN + n * N_HID + k);
    ((uint32_t*)lw2i)[i] = pk2(qi.x, qi.y);
    const float2 qr = *(const float2*)(W2RC + n * N_HID + k);
    ((uint32_t*)lw2r)[i] = pk2(qr.x, qr.y);
  }
  for (int i = tid; i < 2048; i += 256) {
    const int idx = 2 * i;
    const int j = idx & 15, lp = (idx >> 4) & 31, kk = (idx >> 9) & 7;
    const int o = lp & 15;
    const int k = kk * 32 + (lp >> 4) * 16 + j;
    uint32_t u = 0u;
    if (o < N_OUT) { const float2 q = *(const float2*)(W3 + o * N_HID + k); u = pk2(q.x, q.y); }
    ((uint32_t*)lw3)[i] = u;
  }
  for (int i = tid; i < 2048; i += 256) {           // initial spikes are zero
    ((uint32_t*)ls1)[i] = 0u; ((uint32_t*)ls2)[i] = 0u;
  }

  // ---- per-lane neuron parameters (one neuron per lane per n-tile)
  float al1[2], rh1[2], b1v[2], al2[2], rh2[2], b2iv[2], b2rv[2];
  int   sb[2];   // A-layout store base for this lane's neuron column
#pragma unroll
  for (int nt = 0; nt < 2; ++nt) {
    const int n = wave * 32 + nt * 16 + lm;
    al1[nt] = __expf(-1.f / TM1[n]);  rh1[nt] = __expf(-1.f / TAD1[n]);
    al2[nt] = __expf(-1.f / TM2[n]);  rh2[nt] = __expf(-1.f / TAD2[n]);
    b1v[nt] = B1[n]; b2iv[nt] = B2IN[n]; b2rv[nt] = B2RC[n];
    // inverse A-layout: h' = (n>>4)&1, j = n&15
    sb[nt] = (((n >> 5) * 32 + ((n >> 4) & 1) * 16) << 4) + (n & 15);
  }
  float al3 = 0.f, b3v = 0.f;
  if (lm < N_OUT) { al3 = __expf(-1.f / TM3[lm]); b3v = B3[lm]; }

  // ---- t-invariant spikify bookkeeping: this thread's 8 (src offset, validity) pairs
  int xofs[8];
#pragma unroll
  for (int it = 0; it < 8; ++it) {
    const int i = tid + 256 * it;
    const int j = i & 15, lp = (i >> 4) & 31, kk = i >> 9;
    const int m = lp & 15, hh = lp >> 4;
    const int k = kk * 32 + hh * 16 + j;
    if (k < KIN) {
      const int c = k / D_IN, d = k - c * D_IN;
      xofs[it] = (((b0 + m) * C_CH + c) * T_STEPS) * D_IN + d;
    } else xofs[it] = -1;
  }

  // ---- recurrent state in C-fragment layout (row r+8h, col = lane's neuron)
  float m1[2][8], a1[2][8], s1p[2][8], m2[2][8], a2[2][8], s2p[2][8];
#pragma unroll
  for (int nt = 0; nt < 2; ++nt)
#pragma unroll
    for (int r = 0; r < 8; ++r) {
      m1[nt][r] = 0.f; a1[nt][r] = 0.01f; s1p[nt][r] = 0.f;
      m2[nt][r] = 0.f; a2[nt][r] = 0.01f; s2p[nt][r] = 0.f;
    }
  float m3[8], accv[8];
#pragma unroll
  for (int r = 0; r < 8; ++r) { m3[r] = 0.f; accv[r] = 0.f; }

  __syncthreads();

  // ================= sequential scan over time =================
  for (int t = 0; t < T_STEPS; ++t) {
    // spikify straight into A-fragment layout; branchless (clamped load + cndmask)
#pragma unroll
    for (int it = 0; it < 8; ++it) {
      const int of = (xofs[it] >= 0) ? xofs[it] : 0;
      const float xv = X[of + t * D_IN];
      unsigned short bits = (xv - thr > 0.f) ? (unsigned short)0x3F80
                          : (((-thr - xv) > 0.f) ? (unsigned short)0xBF80
                                                 : (unsigned short)0);
      lxt[tid + 256 * it] = (xofs[it] >= 0) ? bits : (unsigned short)0;
    }
    if (t + 1 < T_STEPS && tid < 16)   // global_prefetch_b8 for next step's x lines
      __builtin_prefetch(&X[(((b0 + tid) * C_CH) * T_STEPS + (t + 1)) * D_IN], 0, 1);
    __syncthreads();

    // ---- layer 1: i1 = xs @ W1^T  (K=128 padded; W1 streamed from L2)
    v8f acc1[2];
#pragma unroll
    for (int nt = 0; nt < 2; ++nt)
#pragma unroll
      for (int r = 0; r < 8; ++r) acc1[nt][r] = 0.f;
#pragma unroll
    for (int kk = 0; kk < 4; ++kk) {
      const v16bf af = *(const v16bf*)&lxt[(kk * 32 + lane) << 4];
#pragma unroll
      for (int nt = 0; nt < 2; ++nt) {
        v16bf bw;
        if (SWZ) {   // pre-swizzled bf16 W1 in workspace: one 32B L2-resident load
          bw = *(const v16bf*)&W1S[(((wave * 2 + nt) * 4 + kk) * 32 + lane) << 4];
        } else {
          bw = load_b_w1_stream(W1, wave * 32 + nt * 16 + lm, kk, h);
        }
        acc1[nt] = wmma_bf16(af, bw, acc1[nt]);
      }
    }
    // ALIF 1 + publish s1 (A-fragment layout; spike bits selected, no converts)
#pragma unroll
    for (int nt = 0; nt < 2; ++nt) {
#pragma unroll
      for (int r = 0; r < 8; ++r) {
        const float I = acc1[nt][r] + b1v[nt];
        a1[nt][r] = rh1[nt] * a1[nt][r] + (1.f - rh1[nt]) * s1p[nt][r];
        const float Bth = 0.01f + 1.8f * a1[nt][r];
        m1[nt][r] = al1[nt] * m1[nt][r] + (1.f - al1[nt]) * I - Bth * s1p[nt][r];
        const bool fire = (m1[nt][r] - Bth) > 0.f;
        s1p[nt][r] = fire ? 1.f : 0.f;
        ls1[sb[nt] + ((r + 8 * h) << 4)] = fire ? (unsigned short)0x3F80 : (unsigned short)0;
      }
    }
    __syncthreads();

    // ---- layer 2: i2 = s1 @ W2_in^T + s2(t-1) @ W2_rec^T   (K=256, all LDS)
    v8f acc2[2];
#pragma unroll
    for (int nt = 0; nt < 2; ++nt)
#pragma unroll
      for (int r = 0; r < 8; ++r) acc2[nt][r] = 0.f;
#pragma unroll
    for (int kk = 0; kk < 8; ++kk) {
      const v16bf ai = *(const v16bf*)&ls1[(kk * 32 + lane) << 4];
      const v16bf ar = *(const v16bf*)&ls2[(kk * 32 + lane) << 4];
#pragma unroll
      for (int nt = 0; nt < 2; ++nt) {
        const int ntg = wave * 2 + nt;
        const v16bf bi = *(const v16bf*)&lw2i[((ntg * 8 + kk) * 32 + lane) << 4];
        acc2[nt] = wmma_bf16(ai, bi, acc2[nt]);
        const v16bf br = *(const v16bf*)&lw2r[((ntg * 8 + kk) * 32 + lane) << 4];
        acc2[nt] = wmma_bf16(ar, br, acc2[nt]);
      }
    }
    __syncthreads();   // everyone is done reading s2(t-1)

    // ALIF 2 + publish s2
#pragma unroll
    for (int nt = 0; nt < 2; ++nt) {
#pragma unroll
      for (int r = 0; r < 8; ++r) {
        const float I = acc2[nt][r] + b2iv[nt] + b2rv[nt];
        a2[nt][r] = rh2[nt] * a2[nt][r] + (1.f - rh2[nt]) * s2p[nt][r];
        const float Bth = 0.01f + 1.8f * a2[nt][r];
        m2[nt][r] = al2[nt] * m2[nt][r] + (1.f - al2[nt]) * I - Bth * s2p[nt][r];
        const bool fire = (m2[nt][r] - Bth) > 0.f;
        s2p[nt][r] = fire ? 1.f : 0.f;
        ls2[sb[nt] + ((r + 8 * h) << 4)] = fire ? (unsigned short)0x3F80 : (unsigned short)0;
      }
    }
    __syncthreads();   // s2(t) visible

    // ---- layer 3 readout (wave 0 only; cols 12..15 hit zero-padded W3)
    if (wave == 0) {
      v8f acc3;
#pragma unroll
      for (int r = 0; r < 8; ++r) acc3[r] = 0.f;
#pragma unroll
      for (int kk = 0; kk < 8; ++kk) {
        const v16bf af = *(const v16bf*)&ls2[(kk * 32 + lane) << 4];
        const v16bf bf = *(const v16bf*)&lw3[(kk * 32 + lane) << 4];
        acc3 = wmma_bf16(af, bf, acc3);
      }
#pragma unroll
      for (int r = 0; r < 8; ++r) {
        const float i3 = acc3[r] + b3v;
        m3[r] = al3 * m3[r] + (1.f - al3) * i3;
        accv[r] += m3[r];
      }
    }
  }

  // ================= log_softmax(acc / T) =================
  float* sc = (float*)lxt;                 // reuse input tile as [16][16] f32 scratch
  if (wave == 0) {
#pragma unroll
    for (int r = 0; r < 8; ++r) sc[(r + 8 * h) * 16 + lm] = accv[r];
  }
  __syncthreads();
  if (tid < 16) {
    const int row = tid;
    const float inv_t = 1.f / (float)T_STEPS;
    float mx = -1e30f;
    for (int o = 0; o < N_OUT; ++o) mx = fmaxf(mx, sc[row * 16 + o] * inv_t);
    float se = 0.f;
    for (int o = 0; o < N_OUT; ++o) se += __expf(sc[row * 16 + o] * inv_t - mx);
    const float lse = __logf(se) + mx;
    for (int o = 0; o < N_OUT; ++o)
      OUT[(b0 + row) * N_OUT + o] = sc[row * 16 + o] * inv_t - lse;
  }
}

extern "C" void kernel_launch(void* const* d_in, const int* in_sizes, int n_in,
                              void* d_out, int out_size, void* d_ws, size_t ws_size,
                              hipStream_t stream) {
  (void)in_sizes; (void)n_in; (void)out_size;
  const float* X    = (const float*)d_in[0];
  const float* THR  = (const float*)d_in[1];
  const float* W1   = (const float*)d_in[2];
  const float* B1   = (const float*)d_in[3];
  const float* W2IN = (const float*)d_in[4];
  const float* B2IN = (const float*)d_in[5];
  const float* W2RC = (const float*)d_in[6];
  const float* B2RC = (const float*)d_in[7];
  const float* W3   = (const float*)d_in[8];
  const float* B3   = (const float*)d_in[9];
  const float* TAD1 = (const float*)d_in[10];
  const float* TM1  = (const float*)d_in[11];
  const float* TAD2 = (const float*)d_in[12];
  const float* TM2  = (const float*)d_in[13];
  const float* TM3  = (const float*)d_in[14];
  float* OUT = (float*)d_out;

  if (d_ws != nullptr && ws_size >= (size_t)W1S_BYTES) {
    unsigned short* W1S = (unsigned short*)d_ws;
    w1_swz_kernel<<<dim3((W1S_HALVES / 2 + 255) / 256), dim3(256), 0, stream>>>(W1, W1S);
    alif_rnn_kernel<true><<<dim3(B_TOT / 16), dim3(256), 0, stream>>>(
        X, THR, W1, B1, W2IN, B2IN, W2RC, B2RC, W3, B3,
        TAD1, TM1, TAD2, TM2, TM3, OUT, W1S);
  } else {
    alif_rnn_kernel<false><<<dim3(B_TOT / 16), dim3(256), 0, stream>>>(
        X, THR, W1, B1, W2IN, B2IN, W2RC, B2RC, W3, B3,
        TAD1, TM1, TAD2, TM2, TM3, OUT, nullptr);
  }
}
